// Encoder_63144609186115
// MI455X (gfx1250) — matrix operand order repaired and checked by reading, not compiled, not run on hardware
//
#include <hip/hip_runtime.h>
#include <hip/hip_bf16.h>
#include <stdint.h>

// ---------------------------------------------------------------------------
// 2-layer GRU encoder for gfx1250 (MI455X), wave32 + WMMA bf16.
// Batch-partitioned persistent kernel: 16 workgroups x 16 batch rows, no
// cross-workgroup sync (the recurrence is independent per batch row).
// Weights are packed once to bf16 [N][K] in d_ws and streamed from L2 each
// step (they total ~1.1 MB, L2-resident); state tile lives in LDS.
// ---------------------------------------------------------------------------

typedef __attribute__((ext_vector_type(16))) __bf16 v16bf;
typedef __attribute__((ext_vector_type(8)))  float  v8f;

#define B_    256
#define T_    512
#define VEC_  64
#define H_    256
#define H3_   768
#define BM    16        // batch rows per workgroup
#define NWAVES 24
#define THREADS (NWAVES*32)
#define NT    2         // 48 n-tiles / 24 waves

union U32B { uint4 u[2]; v16bf v; };

// A operand, 16x32 bf16 tile (ISA 7.12.2): lanes 0-15 row m hold K {0..7,16..23},
// lanes 16-31 hold K {8..15,24..31}; two 16-byte chunks 32 bytes apart.
__device__ __forceinline__ v16bf loadA16(const __bf16* row, int kb, int hi) {
  const uint4* p = (const uint4*)row;
  U32B u;
  u.u[0] = p[kb*4 + hi];
  u.u[1] = p[kb*4 + hi + 2];
  return u.v;
}
// B operand, 32x16 bf16 tile: lane holds column n = lane&15, same K order as A
// (lanes 0-15: K {0..7,16..23}, lanes 16-31: K {8..15,24..31}) from [N][K] layout.
__device__ __forceinline__ v16bf loadB16(const __bf16* row, int kb, int hi) {
  const uint4* p = (const uint4*)row;
  int idx = kb*4 + hi;
  U32B u;
  u.u[0] = p[idx];
  u.u[1] = p[idx+2];
  return u.v;
}

__device__ __forceinline__ v8f wmma_bf16(v16bf a, v16bf b, v8f c) {
  return __builtin_amdgcn_wmma_f32_16x16x32_bf16(false, a, false, b,
                                                 (short)0, c, false, false);
}

__device__ __forceinline__ v8f splat8(float x) {
  v8f v;
  #pragma unroll
  for (int j = 0; j < 8; ++j) v[j] = x;
  return v;
}

// -------------------- weight packing (fp32 -> bf16, [K][N] -> [N][K]) ------
__global__ void pack_T(const float* __restrict__ src, __bf16* __restrict__ dst,
                       int K, int N) {
  int i = blockIdx.x*blockDim.x + threadIdx.x;
  if (i >= K*N) return;
  int n = i / K, k = i - n*K;
  dst[i] = (__bf16)src[(size_t)k*N + n];
}
__global__ void pack_C(const float* __restrict__ src, __bf16* __restrict__ dst,
                       int n) {
  int i = blockIdx.x*blockDim.x + threadIdx.x;
  if (i < n) dst[i] = (__bf16)src[i];
}

// -------------------- persistent GRU kernel --------------------------------
__global__ void __launch_bounds__(THREADS)
gru_persistent(const int* __restrict__ tok,
               const float* __restrict__ h0,
               const float* __restrict__ b1_, const float* __restrict__ b2_,
               const __bf16* __restrict__ U1T, const __bf16* __restrict__ U2T,
               const __bf16* __restrict__ W1T, const __bf16* __restrict__ W2T,
               const __bf16* __restrict__ embB,
               float* __restrict__ out)
{
  extern __shared__ char smem[];
  float*  hu_l  = (float*)(smem);               // [16][768] h@U + b1
  float*  xw_l  = (float*)(smem + 49152);       // [16][768] x@W + b0
  float*  h_f   = (float*)(smem + 98304);       // [16][256] fp32 state
  __bf16* h_bf  = (__bf16*)(smem + 114688);     // [16][256] bf16 state (A op)
  __bf16* x_bf  = (__bf16*)(smem + 122880);     // [16][256] bf16 layer-2 input
  float*  out_f = (float*)(smem + 131072);      // [16][256] Keras out carry

  const int tid  = threadIdx.x;
  const int lane = tid & 31;
  const int wave = tid >> 5;
  const int m    = lane & 15;
  const int hi   = lane >> 4;
  const int bb   = blockIdx.x * BM;

  float* outx = out;                            // x: [B][T][2H]
  float* outh = out + (size_t)B_*T_*2*H_;       // h: [B][2H]

  const int col0 = (wave*NT + 0)*16 + m;        // this wave's two N columns
  const int col1 = (wave*NT + 1)*16 + m;

  for (int layer = 0; layer < 2; ++layer) {
    const __bf16* UT = layer ? U2T : U1T;
    const __bf16* WT = layer ? W2T : W1T;
    const float*  bs = layer ? b2_ : b1_;

    // Initialize state / output-carry for this layer.
    for (int e = tid; e < BM*H_; e += THREADS) {
      int mm = e >> 8, j = e & 255;
      float hv = h0[(size_t)(bb+mm)*H_ + j];
      h_f[e]   = hv;
      h_bf[e]  = (__bf16)hv;
      out_f[e] = 0.f;
    }
    __syncthreads();

    for (int t = 0; t < T_; ++t) {
      // Stage layer-2 input x_lr[:,t,:] (fp32 in d_out) as bf16 in LDS.
      if (layer) {
        for (int e = tid; e < BM*H_; e += THREADS) {
          int mm = e >> 8, j = e & 255;
          size_t ix = ((size_t)(bb+mm)*T_ + t)*(2*H_) + j;
          x_bf[e] = (__bf16)outx[ix];
          if (t + 1 < T_) __builtin_prefetch(outx + ix + 2*H_, 0, 3);
        }
      }
      __syncthreads();

      // ---- per-step GEMMs: acc = bias; acc += A(16xK) * B(Kx16) -----------
      v8f axw0 = splat8(bs[col0]);
      v8f axw1 = splat8(bs[col1]);
      v8f ahu0 = splat8(bs[H3_ + col0]);
      v8f ahu1 = splat8(bs[H3_ + col1]);

      if (layer == 0) {
        // x@W, K=64, A gathered straight from the bf16 embedding table.
        int tokin = tok[(size_t)(bb+m)*T_ + (T_-1-t)];
        const __bf16* erow = embB + (size_t)tokin*VEC_;
        const __bf16* w0 = WT + (size_t)col0*VEC_;
        const __bf16* w1 = WT + (size_t)col1*VEC_;
        #pragma unroll
        for (int kb = 0; kb < 2; ++kb) {
          v16bf a = loadA16(erow, kb, hi);        // global_load_b128
          axw0 = wmma_bf16(a, loadB16(w0, kb, hi), axw0);
          axw1 = wmma_bf16(a, loadB16(w1, kb, hi), axw1);
        }
      } else {
        // x@W, K=256, A from LDS-staged bf16 x_lr tile.
        const __bf16* xr = x_bf + m*H_;
        const __bf16* w0 = WT + (size_t)col0*H_;
        const __bf16* w1 = WT + (size_t)col1*H_;
        #pragma unroll
        for (int kb = 0; kb < 8; ++kb) {
          v16bf a = loadA16(xr, kb, hi);          // ds_load_b128
          axw0 = wmma_bf16(a, loadB16(w0, kb, hi), axw0);
          axw1 = wmma_bf16(a, loadB16(w1, kb, hi), axw1);
        }
      }

      // h@U, K=256, A from LDS state tile, B streamed from L2.
      {
        const __bf16* hr = h_bf + m*H_;
        const __bf16* u0 = UT + (size_t)col0*H_;
        const __bf16* u1 = UT + (size_t)col1*H_;
        #pragma unroll
        for (int kb = 0; kb < 8; ++kb) {
          v16bf a = loadA16(hr, kb, hi);          // ds_load_b128
          ahu0 = wmma_bf16(a, loadB16(u0, kb, hi), ahu0);
          ahu1 = wmma_bf16(a, loadB16(u1, kb, hi), ahu1);
        }
      }

      // C/D layout: c[j] = element (row j + 8*hi, col)
      #pragma unroll
      for (int j = 0; j < 8; ++j) {
        int row = j + hi*8;
        xw_l[row*H3_ + col0] = axw0[j];
        xw_l[row*H3_ + col1] = axw1[j];
        hu_l[row*H3_ + col0] = ahu0[j];
        hu_l[row*H3_ + col1] = ahu1[j];
      }
      __syncthreads();

      // Gates + masked state update (Keras reset_after=True, order z,r,h).
      for (int e = tid; e < BM*H_; e += THREADS) {
        int mm = e >> 8, j = e & 255;
        int tm = layer ? (T_-1-t) : t;            // mask_lr vs mask_rl
        bool msk = tok[(size_t)(bb+mm)*T_ + tm] != 0;
        float xz = xw_l[mm*H3_ + j],          hz = hu_l[mm*H3_ + j];
        float xr = xw_l[mm*H3_ + H_ + j],     hr = hu_l[mm*H3_ + H_ + j];
        float xh = xw_l[mm*H3_ + 2*H_ + j],   hh = hu_l[mm*H3_ + 2*H_ + j];
        float z  = 1.f/(1.f + __expf(-(xz+hz)));
        float r  = 1.f/(1.f + __expf(-(xr+hr)));
        float hc = tanhf(xh + r*hh);
        float hold  = h_f[e];
        float hcand = z*hold + (1.f - z)*hc;
        float hnew  = msk ? hcand : hold;
        float onew  = msk ? hcand : out_f[e];
        h_f[e]   = hnew;
        h_bf[e]  = (__bf16)hnew;
        out_f[e] = onew;
        int tt = layer ? (T_-1-t) : t;            // layer2 output re-reversed
        outx[((size_t)(bb+mm)*T_ + tt)*(2*H_) + layer*H_ + j] = onew;
      }
      __syncthreads();
    }

    // Final carried state h_T.
    for (int e = tid; e < BM*H_; e += THREADS) {
      int mm = e >> 8, j = e & 255;
      outh[(size_t)(bb+mm)*(2*H_) + layer*H_ + j] = h_f[e];
    }
    __threadfence();     // make layer-1 x visible to this WG's layer-2 reads
    __syncthreads();
  }
}

// ---------------------------------------------------------------------------
extern "C" void kernel_launch(void* const* d_in, const int* in_sizes, int n_in,
                              void* d_out, int out_size, void* d_ws, size_t ws_size,
                              hipStream_t stream) {
  (void)in_sizes; (void)n_in; (void)out_size; (void)ws_size;
  const int*   tok = (const int*)  d_in[0];
  const float* h0  = (const float*)d_in[1];
  const float* emb = (const float*)d_in[3];
  const float* Wlr = (const float*)d_in[4];
  const float* Ulr = (const float*)d_in[5];
  const float* blr = (const float*)d_in[6];
  const float* Wrl = (const float*)d_in[7];
  const float* Url = (const float*)d_in[8];
  const float* brl = (const float*)d_in[9];

  char* ws = (char*)d_ws;                     // ~1.26 MB of packed weights
  __bf16* U1T  = (__bf16*)(ws + 0);           // [768][256]
  __bf16* U2T  = (__bf16*)(ws + 393216);      // [768][256]
  __bf16* W2T  = (__bf16*)(ws + 786432);      // [768][256]
  __bf16* W1T  = (__bf16*)(ws + 1179648);     // [768][64]
  __bf16* embB = (__bf16*)(ws + 1277952);     // [104][64]

  pack_T<<<(256*768+255)/256, 256, 0, stream>>>(Ulr, U1T, 256, 768);
  pack_T<<<(256*768+255)/256, 256, 0, stream>>>(Url, U2T, 256, 768);
  pack_T<<<(256*768+255)/256, 256, 0, stream>>>(Wrl, W2T, 256, 768);
  pack_T<<<( 64*768+255)/256, 256, 0, stream>>>(Wlr, W1T,  64, 768);
  pack_C<<<(104*64 +255)/256, 256, 0, stream>>>(emb, embB, 104*64);

  gru_persistent<<<B_/BM, THREADS, 147456, stream>>>(
      tok, h0, blr, brl, U1T, U2T, W1T, W2T, embB, (float*)d_out);
}
